// GCN_syn2_3375844295347
// MI455X (gfx1250) — matrix operand order repaired and checked by reading, not compiled, not run on hardware
//
#include <hip/hip_runtime.h>

typedef __attribute__((ext_vector_type(16))) _Float16 v16h;
typedef __attribute__((ext_vector_type(8)))  float    v8f;

#define HID  20
#define NCLS 10
#define KPAD 32   // padded K / lda for hidden-layer GEMM input

// ---------------- utility kernels ----------------

__global__ void k_zero(float* __restrict__ p, long n) {
  long i = (long)blockIdx.x * blockDim.x + threadIdx.x;
  if (i < n) p[i] = 0.0f;
}

__global__ void k_deg(const int* __restrict__ col, float* __restrict__ deg, int E) {
  int e = blockIdx.x * blockDim.x + threadIdx.x;
  if (e < E) atomicAdd(&deg[col[e]], 1.0f);
}

__global__ void k_dinv(float* __restrict__ deg, int N) {
  int i = blockIdx.x * blockDim.x + threadIdx.x;
  if (i < N) {
    float d = deg[i];
    deg[i] = (d > 0.0f) ? rsqrtf(d) : 0.0f;
  }
}

__global__ void k_norm(const int* __restrict__ row, const int* __restrict__ col,
                       const float* __restrict__ dinv, float* __restrict__ norm, int E) {
  int e = blockIdx.x * blockDim.x + threadIdx.x;
  if (e < E) norm[e] = dinv[row[e]] * dinv[col[e]];
}

// ---------------- B-fragment pre-pack ----------------
// Convert B[K x 20] (f32) into WMMA-ready f16 fragments, zero-padded.
// Layout: Bp[(((ks*2 + ntile)*32 + lane)*16 + e]
//   lane -> (n = ntile*16 + (lane&15)), e -> (k = ks*32 + (lane>>4)*16 + e)
// GEMM loads one fragment as a single aligned 32-byte v16h per lane.

__global__ void k_prepB(const float* __restrict__ B, int K, int ksteps,
                        _Float16* __restrict__ Bp) {
  int idx = blockIdx.x * blockDim.x + threadIdx.x;
  int total = ksteps * 2 * 32 * 16;
  if (idx >= total) return;
  int e  = idx & 15;
  int l  = (idx >> 4) & 31;
  int t  = (idx >> 9) & 1;
  int ks = idx >> 10;
  int k  = ks * 32 + (l >> 4) * 16 + e;
  int n  = t * 16 + (l & 15);
  float v = (k < K && n < HID) ? B[k * HID + n] : 0.0f;
  Bp[idx] = (_Float16)v;
}

// ---------------- WMMA GEMM: T[N,20] = A[N,Kpad](lda) @ B ----------------
// One wave per 16-row tile, two 16x16x32 f16 WMMAs per k-step (f32 accum).
// No bounds checks in the loop: row index clamped once, K padded to 32.

__global__ void k_gemm_wmma(const float* __restrict__ A, int lda, int ksteps,
                            const _Float16* __restrict__ Bp,
                            float* __restrict__ T, int N) {
  const int lane = threadIdx.x & 31;
  const int wave = threadIdx.x >> 5;
  const int tile = blockIdx.x * (blockDim.x >> 5) + wave;
  const int m    = lane & 15;
  const int hi   = lane >> 4;
  const int rowbase = tile * 16;
  const int r  = rowbase + m;
  const int rc = (r < N) ? r : (N - 1);        // clamp: garbage rows never stored

  const float* __restrict__ ar = A + (long)rc * lda;

  v8f c0 = {};
  v8f c1 = {};

  for (int ks = 0; ks < ksteps; ++ks) {
    const int k0 = ks << 5;
    const int ka = k0 + hi * 16;       // first contiguous 8-float run
    const int kb = k0 + hi * 16 + 8;   // second contiguous 8-float run

    float4 f0 = *(const float4*)(ar + ka);
    float4 f1 = *(const float4*)(ar + ka + 4);
    float4 f2 = *(const float4*)(ar + kb);
    float4 f3 = *(const float4*)(ar + kb + 4);

    v16h a;
    a[0]  = (_Float16)f0.x; a[1]  = (_Float16)f0.y;
    a[2]  = (_Float16)f0.z; a[3]  = (_Float16)f0.w;
    a[4]  = (_Float16)f1.x; a[5]  = (_Float16)f1.y;
    a[6]  = (_Float16)f1.z; a[7]  = (_Float16)f1.w;
    a[8]  = (_Float16)f2.x; a[9]  = (_Float16)f2.y;
    a[10] = (_Float16)f2.z; a[11] = (_Float16)f2.w;
    a[12] = (_Float16)f3.x; a[13] = (_Float16)f3.y;
    a[14] = (_Float16)f3.z; a[15] = (_Float16)f3.w;

    v16h b0 = *(const v16h*)(Bp + (((ks * 2 + 0) * 32 + lane) << 4));
    v16h b1 = *(const v16h*)(Bp + (((ks * 2 + 1) * 32 + lane) << 4));

    c0 = __builtin_amdgcn_wmma_f32_16x16x32_f16(false, a, false, b0, (short)0, c0, false, false);
    c1 = __builtin_amdgcn_wmma_f32_16x16x32_f16(false, a, false, b1, (short)0, c1, false, false);
  }

  // C/D layout: vgpr rr, lane: M = rr + 8*hi, N = lane&15
#pragma unroll
  for (int rr = 0; rr < 8; ++rr) {
    int orow = rowbase + rr + (hi ? 8 : 0);
    if (orow < N) {
      T[(long)orow * HID + m] = c0[rr];
      if (m < HID - 16) T[(long)orow * HID + 16 + m] = c1[rr];
    }
  }
}

// ---------------- edge scatter: agg[col] += T[row] * norm ----------------
// One thread per (edge, feature); contiguous 20-float gather + atomic burst
// resolving in L2 (agg is 8 MB << 192 MB L2).

__global__ void k_scatter(const int* __restrict__ row, const int* __restrict__ col,
                          const float* __restrict__ norm, const float* __restrict__ T,
                          float* __restrict__ agg, unsigned int EH) {
  unsigned int t = blockIdx.x * blockDim.x + threadIdx.x;
  if (t >= EH) return;
  unsigned int e = t / HID;
  unsigned int f = t - e * HID;
  float v = T[(long)row[e] * HID + f] * norm[e];
  atomicAdd(&agg[(long)col[e] * HID + f], v);
}

// ---------------- bias + ReLU -> concat slice + zero-padded GEMM input ----------------

__global__ void k_biasrelu(const float* __restrict__ agg, const float* __restrict__ b,
                           float* __restrict__ hcat, int coloff,
                           float* __restrict__ hpad, unsigned int NH) {
  unsigned int t = blockIdx.x * blockDim.x + threadIdx.x;
  if (t >= NH) return;
  unsigned int i = t / HID;
  unsigned int f = t - i * HID;
  float v = fmaxf(agg[t] + b[f], 0.0f);
  hcat[(long)i * (3 * HID) + coloff + f] = v;
  hpad[(long)i * KPAD + f] = v;          // cols 20..31 pre-zeroed once
}

// ---------------- head: logits = hcat @ Wp + bp ; log_softmax ----------------

__global__ void k_head(const float* __restrict__ hcat, const float* __restrict__ Wp,
                       const float* __restrict__ bp, float* __restrict__ out, int N) {
  __shared__ float sW[3 * HID * NCLS];
  __shared__ float sb[NCLS];
  for (int j = threadIdx.x; j < 3 * HID * NCLS; j += blockDim.x) sW[j] = Wp[j];
  if (threadIdx.x < NCLS) sb[threadIdx.x] = bp[threadIdx.x];
  __syncthreads();

  int i = blockIdx.x * blockDim.x + threadIdx.x;
  if (i >= N) return;

  float acc[NCLS];
#pragma unroll
  for (int c = 0; c < NCLS; ++c) acc[c] = sb[c];

  const float* hr = hcat + (long)i * (3 * HID);
#pragma unroll 4
  for (int k = 0; k < 3 * HID; ++k) {
    float hv = hr[k];
#pragma unroll
    for (int c = 0; c < NCLS; ++c) acc[c] += hv * sW[k * NCLS + c];
  }

  float mx = acc[0];
#pragma unroll
  for (int c = 1; c < NCLS; ++c) mx = fmaxf(mx, acc[c]);
  float s = 0.0f;
#pragma unroll
  for (int c = 0; c < NCLS; ++c) s += expf(acc[c] - mx);
  float lse = logf(s) + mx;
#pragma unroll
  for (int c = 0; c < NCLS; ++c) out[(long)i * NCLS + c] = acc[c] - lse;
}

// ---------------- launcher ----------------

extern "C" void kernel_launch(void* const* d_in, const int* in_sizes, int n_in,
                              void* d_out, int out_size, void* d_ws, size_t ws_size,
                              hipStream_t stream) {
  const float* x  = (const float*)d_in[0];
  const int*   ei = (const int*)  d_in[1];
  const float* W1 = (const float*)d_in[2];
  const float* b1 = (const float*)d_in[3];
  const float* W2 = (const float*)d_in[4];
  const float* b2 = (const float*)d_in[5];
  const float* W3 = (const float*)d_in[6];
  const float* b3 = (const float*)d_in[7];
  const float* Wp = (const float*)d_in[8];
  const float* bp = (const float*)d_in[9];

  const int N = in_sizes[0] / 128;
  const int E = in_sizes[1] / 2;
  const int* row = ei;       // edge_index[0] = src
  const int* col = ei + E;   // edge_index[1] = dst

  // workspace carve-out:
  // dinv[N] | norm[E] | T[20N] | agg[20N] | hcat[60N] | hpad[32N] | Bp1/2/3 (f16)
  float* ws   = (float*)d_ws;
  float* dinv = ws;
  float* norm = dinv + N;
  float* T    = norm + E;
  float* agg  = T + (long)HID * N;
  float* hcat = agg + (long)HID * N;
  float* hpad = hcat + (long)3 * HID * N;
  _Float16* Bp1 = (_Float16*)(hpad + (long)KPAD * N);  // 4 ksteps -> 4096 halves
  _Float16* Bp2 = Bp1 + 4096;                          // 1 kstep  -> 1024 halves
  _Float16* Bp3 = Bp2 + 1024;

  const int B = 256;
  auto cdiv = [](long a, long b) { return (int)((a + b - 1) / b); };

  const long NHl = (long)N * HID;
  const unsigned int NH = (unsigned int)NHl;
  const unsigned int EH = (unsigned int)E * HID;
  const int tiles      = (N + 15) / 16;
  const int gemmBlocks = (tiles + 7) / 8;   // 8 waves / block

  // edge normalization (layer-invariant, computed once)
  k_zero<<<cdiv(N, B), B, 0, stream>>>(dinv, N);
  k_deg <<<cdiv(E, B), B, 0, stream>>>(col, dinv, E);
  k_dinv<<<cdiv(N, B), B, 0, stream>>>(dinv, N);
  k_norm<<<cdiv(E, B), B, 0, stream>>>(row, col, dinv, norm, E);

  // zero hpad once (pad columns stay zero across layers)
  k_zero<<<cdiv((long)KPAD * N, B), B, 0, stream>>>(hpad, (long)KPAD * N);

  // pre-pack weights into WMMA B-fragment layout (f16)
  k_prepB<<<cdiv(4 * 1024, B), B, 0, stream>>>(W1, 128, 4, Bp1);
  k_prepB<<<cdiv(1 * 1024, B), B, 0, stream>>>(W2, HID, 1, Bp2);
  k_prepB<<<cdiv(1 * 1024, B), B, 0, stream>>>(W3, HID, 1, Bp3);

  // layer 1: x[N,128] @ W1
  k_gemm_wmma<<<gemmBlocks, B, 0, stream>>>(x, 128, 4, Bp1, T, N);
  k_zero     <<<cdiv(NHl, B), B, 0, stream>>>(agg, NHl);
  k_scatter  <<<cdiv(EH, B), B, 0, stream>>>(row, col, norm, T, agg, EH);
  k_biasrelu <<<cdiv(NHl, B), B, 0, stream>>>(agg, b1, hcat, 0, hpad, NH);

  // layer 2: h1 (hpad, K padded to 32) @ W2
  k_gemm_wmma<<<gemmBlocks, B, 0, stream>>>(hpad, KPAD, 1, Bp2, T, N);
  k_zero     <<<cdiv(NHl, B), B, 0, stream>>>(agg, NHl);
  k_scatter  <<<cdiv(EH, B), B, 0, stream>>>(row, col, norm, T, agg, EH);
  k_biasrelu <<<cdiv(NHl, B), B, 0, stream>>>(agg, b2, hcat, HID, hpad, NH);

  // layer 3: h2 (hpad) @ W3
  k_gemm_wmma<<<gemmBlocks, B, 0, stream>>>(hpad, KPAD, 1, Bp3, T, N);
  k_zero     <<<cdiv(NHl, B), B, 0, stream>>>(agg, NHl);
  k_scatter  <<<cdiv(EH, B), B, 0, stream>>>(row, col, norm, T, agg, EH);
  k_biasrelu <<<cdiv(NHl, B), B, 0, stream>>>(agg, b3, hcat, 2 * HID, hpad, NH);

  // head + log_softmax
  k_head<<<cdiv(N, B), B, 0, stream>>>(hcat, Wp, bp, (float*)d_out, N);
}
